// GenerateXs_10969346474265
// MI455X (gfx1250) — matrix operand order repaired and checked by reading, not compiled, not run on hardware
//
#include <hip/hip_runtime.h>
#include <hip/hip_bf16.h>
#include <math.h>

// Problem constants (match reference setup_inputs)
#define SEQ   512
#define BATCH 64
#define HID   1024
#define NUMX  8     // num_xs is a device scalar; cannot sync-read under graph capture

typedef __attribute__((ext_vector_type(16))) __bf16       v16bf;
typedef __attribute__((ext_vector_type(8)))  float        v8f;
typedef __attribute__((ext_vector_type(4)))  unsigned int u32x4;
typedef __attribute__((ext_vector_type(2)))  unsigned int u32x2;

union FragBF { v16bf v; u32x4 q[2]; };
union Pk8    { u32x4 q; __bf16 e[8]; };
union Pk4    { u32x2 d; __bf16 e[4]; };
union Pk2    { unsigned int u; __bf16 e[2]; };

// ---------------------------------------------------------------------------
// Convert weight matrices f32 -> bf16 (rows stay K-contiguous). Vectorized 4-wide.
// ---------------------------------------------------------------------------
__global__ void cvt_w_kernel(const float* __restrict__ WK, const float* __restrict__ WV,
                             __bf16* __restrict__ WKb, __bf16* __restrict__ WVb) {
    int i = (blockIdx.x * blockDim.x + threadIdx.x) * 4;   // < HID*HID
    float4 k = *(const float4*)&WK[i];
    float4 v = *(const float4*)&WV[i];
    Pk4 pk, pv;
    pk.e[0] = (__bf16)k.x; pk.e[1] = (__bf16)k.y; pk.e[2] = (__bf16)k.z; pk.e[3] = (__bf16)k.w;
    pv.e[0] = (__bf16)v.x; pv.e[1] = (__bf16)v.y; pv.e[2] = (__bf16)v.z; pv.e[3] = (__bf16)v.w;
    *(u32x2*)&WKb[i] = pk.d;
    *(u32x2*)&WVb[i] = pv.d;
}

// ---------------------------------------------------------------------------
// Fused K/V projection GEMM:
//   C[m, n] = sum_h A[m,h] * W[n,h] + bias[n],  m = b*SEQ + s, A row = hidden[s][b][:]
// Block: 256 threads (8 waves). Tile 64(M) x 64(N), BK = 64 (two WMMA k-halves).
// Waves 0..3 produce K rows, waves 4..7 produce V (shared A tile in LDS).
// bf16 WMMA 16x16x32, f32 accumulate; epilogue bounces through LDS for b128 stores.
// ---------------------------------------------------------------------------
__global__ void __launch_bounds__(256)
proj_kv_kernel(const float*  __restrict__ hidden,
               const __bf16* __restrict__ WKb, const __bf16* __restrict__ WVb,
               const float*  __restrict__ bK,  const float*  __restrict__ bV,
               __bf16* __restrict__ Kb, __bf16* __restrict__ Vb) {
    constexpr int BM = 64, BN = 64, BKc = 64, LDP = 72;  // 144B pitch: 16B aligned, bank-spread
    __shared__ __align__(16) __bf16 lsA[BM][LDP];
    __shared__ __align__(16) __bf16 lsK[BN][LDP];
    __shared__ __align__(16) __bf16 lsV[BN][LDP];

    const int tid  = threadIdx.x;
    const int lane = tid & 31;
    const int wid  = tid >> 5;
    const int mat  = wid >> 2;           // 0 = K projection, 1 = V projection
    const int rblk = wid & 3;            // 16-row block within the 64-row tile
    const int m0   = blockIdx.y * BM;
    const int n0   = blockIdx.x * BN;

    v8f acc0 = {}, acc1 = {}, acc2 = {}, acc3 = {};

    for (int k0 = 0; k0 < HID; k0 += BKc) {
        // ---- Stage A tile (64x64 f32 -> bf16): float4 loads, b64 LDS stores ----
        #pragma unroll
        for (int it = 0; it < 4; ++it) {
            int i  = tid + it * 256;                 // < 1024
            int r  = i >> 4;
            int c4 = (i & 15) << 2;
            int m  = m0 + r;
            int b  = m >> 9;                         // m / SEQ
            int s  = m & (SEQ - 1);
            float4 f = *(const float4*)&hidden[((size_t)s * BATCH + b) * HID + k0 + c4];
            Pk4 p;
            p.e[0] = (__bf16)f.x; p.e[1] = (__bf16)f.y;
            p.e[2] = (__bf16)f.z; p.e[3] = (__bf16)f.w;
            *(u32x2*)&lsA[r][c4] = p.d;
        }
        // ---- Stage both weight tiles (64x64 bf16): b128 load + b128 LDS store ----
        #pragma unroll
        for (int it = 0; it < 2; ++it) {
            int i  = tid + it * 256;                 // < 512
            int r  = i >> 3;
            int c8 = (i & 7) << 3;
            size_t g = (size_t)(n0 + r) * HID + k0 + c8;
            *(u32x4*)&lsK[r][c8] = *(const u32x4*)&WKb[g];
            *(u32x4*)&lsV[r][c8] = *(const u32x4*)&WVb[g];
        }
        __syncthreads();

        const __bf16 (*lsW)[LDP] = (mat == 0) ? lsK : lsV;
        #pragma unroll
        for (int kh = 0; kh < 2; ++kh) {
            // A fragment (16x32 bf16): lane L -> M = L%16; K = 16*(L/16)+0..15.
            // Two 16B LDS loads.
            FragBF a;
            {
                int row = rblk * 16 + (lane & 15);
                int kb  = kh * 32 + ((lane >> 4) << 4);
                a.q[0] = *(const u32x4*)&lsA[row][kb];
                a.q[1] = *(const u32x4*)&lsA[row][kb + 8];
            }
            // B fragment (32x16 = W^T tile): lane L -> N = L%16, K = 16*(L/16)+0..15
            // == 32 contiguous bytes of one W row.
            const int col = lane & 15;
            const int kk  = kh * 32 + ((lane >> 4) << 4);
            FragBF b0, b1, b2, b3;
            b0.q[0] = *(const u32x4*)&lsW[col     ][kk];  b0.q[1] = *(const u32x4*)&lsW[col     ][kk + 8];
            b1.q[0] = *(const u32x4*)&lsW[col + 16][kk];  b1.q[1] = *(const u32x4*)&lsW[col + 16][kk + 8];
            b2.q[0] = *(const u32x4*)&lsW[col + 32][kk];  b2.q[1] = *(const u32x4*)&lsW[col + 32][kk + 8];
            b3.q[0] = *(const u32x4*)&lsW[col + 48][kk];  b3.q[1] = *(const u32x4*)&lsW[col + 48][kk + 8];

            acc0 = __builtin_amdgcn_wmma_f32_16x16x32_bf16(false, a.v, false, b0.v, (short)0, acc0, false, false);
            acc1 = __builtin_amdgcn_wmma_f32_16x16x32_bf16(false, a.v, false, b1.v, (short)0, acc1, false, false);
            acc2 = __builtin_amdgcn_wmma_f32_16x16x32_bf16(false, a.v, false, b2.v, (short)0, acc2, false, false);
            acc3 = __builtin_amdgcn_wmma_f32_16x16x32_bf16(false, a.v, false, b3.v, (short)0, acc3, false, false);
        }
        __syncthreads();
    }

    // ---- Epilogue: C/D layout VGPR i -> M = i + 8*(lane/16), N = lane%16.
    // Add bias, stash bf16 tile in LDS (reuse lsA), then b128 coalesced global stores.
    const float* bias = (mat == 0) ? bK : bV;
    const int rloc  = rblk * 16 + 8 * (lane >> 4);
    const int cloc  = lane & 15;

    auto stash = [&](const v8f& acc, int j) {
        int   c  = cloc + j * 16;
        float bv = bias[n0 + c];
        #pragma unroll
        for (int i = 0; i < 8; ++i)
            lsA[rloc + i][c] = (__bf16)(acc[i] + bv);
    };
    auto copy_out = [&](__bf16* __restrict__ dst) {
        #pragma unroll
        for (int it = 0; it < 2; ++it) {
            int i  = tid + it * 256;                 // < 512
            int r  = i >> 3;
            int c8 = (i & 7) << 3;
            *(u32x4*)&dst[(size_t)(m0 + r) * HID + n0 + c8] = *(const u32x4*)&lsA[r][c8];
        }
    };

    if (mat == 0) { stash(acc0, 0); stash(acc1, 1); stash(acc2, 2); stash(acc3, 3); }
    __syncthreads();
    copy_out(Kb);
    __syncthreads();
    if (mat == 1) { stash(acc0, 0); stash(acc1, 1); stash(acc2, 2); stash(acc3, 3); }
    __syncthreads();
    copy_out(Vb);
}

// ---------------------------------------------------------------------------
// x0 = problem_q; also write output slot t = 0.
// ---------------------------------------------------------------------------
__global__ void init_kernel(const float* __restrict__ q,
                            float* __restrict__ x, float* __restrict__ out) {
    int i = blockIdx.x * blockDim.x + threadIdx.x;   // < BATCH*HID
    float v = q[i];
    x[i] = v;
    int b = i >> 10, h = i & (HID - 1);
    out[((size_t)b * NUMX) * HID + h] = v;
}

// ---------------------------------------------------------------------------
// scores[b,s] = sum_h x[b,h]*K[b,s,h]; softmax over s. One 512-thread block / batch.
// Each wave owns 32 rows; 32 lanes read 8 bf16 each via b128 (coalesced 512B/wave),
// shuffle-reduce. x[b,:] cached in LDS.
// ---------------------------------------------------------------------------
__global__ void __launch_bounds__(512)
scores_softmax_kernel(const __bf16* __restrict__ Kb, const float* __restrict__ x,
                      float* __restrict__ attn) {
    const int b = blockIdx.x, tid = threadIdx.x;
    const int lane = tid & 31, wid = tid >> 5;
    __shared__ float lx[HID];
    __shared__ float sc[SEQ];
    __shared__ float red[512];

    lx[tid]       = x[(size_t)b * HID + tid];
    lx[tid + 512] = x[(size_t)b * HID + tid + 512];
    __syncthreads();

    for (int r = 0; r < 32; ++r) {
        int s = wid * 32 + r;
        const __bf16* krow = Kb + ((size_t)b * SEQ + s) * HID;
        float p = 0.f;
        #pragma unroll
        for (int i = 0; i < 4; ++i) {
            int h0 = i * 256 + lane * 8;
            Pk8 kv;
            kv.q = *(const u32x4*)&krow[h0];
            #pragma unroll
            for (int j = 0; j < 8; ++j) p += (float)kv.e[j] * lx[h0 + j];
        }
        #pragma unroll
        for (int off = 16; off > 0; off >>= 1) p += __shfl_down(p, off);
        if (lane == 0) sc[s] = p;
    }
    __syncthreads();

    float v = sc[tid];
    red[tid] = v;
    __syncthreads();
    for (int st = 256; st > 0; st >>= 1) {
        if (tid < st) red[tid] = fmaxf(red[tid], red[tid + st]);
        __syncthreads();
    }
    float mx = red[0];
    __syncthreads();
    float e = __expf(v - mx);
    red[tid] = e;
    __syncthreads();
    for (int st = 256; st > 0; st >>= 1) {
        if (tid < st) red[tid] += red[tid + st];
        __syncthreads();
    }
    attn[(size_t)b * SEQ + tid] = e * (1.f / red[0]);
}

// ---------------------------------------------------------------------------
// nx[b,h] = sum_s attn[b,s]*V[b,s,h]. Grid (2, BATCH); each thread owns 2 adjacent h
// (packed b32 V loads, coalesced 1KB/block/iter). attn cached in LDS.
// Writes next x and output slot t+1 as float2.
// ---------------------------------------------------------------------------
__global__ void __launch_bounds__(256)
attend_v_kernel(const __bf16* __restrict__ Vb, const float* __restrict__ attn,
                float* __restrict__ x, float* __restrict__ out, int t) {
    const int b  = blockIdx.y;
    const int h0 = blockIdx.x * 512 + threadIdx.x * 2;
    __shared__ float la[SEQ];
    la[threadIdx.x]       = attn[(size_t)b * SEQ + threadIdx.x];
    la[threadIdx.x + 256] = attn[(size_t)b * SEQ + threadIdx.x + 256];
    __syncthreads();

    const __bf16* vp = Vb + (size_t)b * SEQ * HID + h0;
    float a0 = 0.f, a1 = 0.f;
    for (int s = 0; s < SEQ; ++s) {
        Pk2 vv;
        vv.u = *(const unsigned int*)&vp[(size_t)s * HID];
        float w = la[s];
        a0 += w * (float)vv.e[0];
        a1 += w * (float)vv.e[1];
    }
    float2 r = make_float2(a0, a1);
    *(float2*)&x[(size_t)b * HID + h0] = r;
    *(float2*)&out[((size_t)b * NUMX + (t + 1)) * HID + h0] = r;
}

// ---------------------------------------------------------------------------
extern "C" void kernel_launch(void* const* d_in, const int* in_sizes, int n_in,
                              void* d_out, int out_size, void* d_ws, size_t ws_size,
                              hipStream_t stream) {
    // inputs: 0:num_xs(int) 1:hidden[S,B,H] 2:problem_q[B,H] 3:WK[H,H] 4:bK[H] 5:WV[H,H] 6:bV[H]
    const float* hidden = (const float*)d_in[1];
    const float* q      = (const float*)d_in[2];
    const float* WK     = (const float*)d_in[3];
    const float* bKp    = (const float*)d_in[4];
    const float* WV     = (const float*)d_in[5];
    const float* bVp    = (const float*)d_in[6];
    float* out = (float*)d_out;

    // Workspace carve (~132.5 MiB): bf16 weights, bf16 K/V (fit in 192MB L2), attn, x.
    char* ws = (char*)d_ws;
    size_t off = 0;
    auto carve = [&](size_t bytes) -> void* {
        void* p = ws + off;
        off += (bytes + 255) & ~(size_t)255;
        return p;
    };
    __bf16* WKb = (__bf16*)carve((size_t)HID * HID * 2);
    __bf16* WVb = (__bf16*)carve((size_t)HID * HID * 2);
    __bf16* Kb  = (__bf16*)carve((size_t)BATCH * SEQ * HID * 2);
    __bf16* Vb  = (__bf16*)carve((size_t)BATCH * SEQ * HID * 2);
    float*  attn = (float*)carve((size_t)BATCH * SEQ * 4);
    float*  x    = (float*)carve((size_t)BATCH * HID * 4);

    cvt_w_kernel<<<HID * HID / (256 * 4), 256, 0, stream>>>(WK, WV, WKb, WVb);
    proj_kv_kernel<<<dim3(HID / 64, BATCH * SEQ / 64), 256, 0, stream>>>(
        hidden, WKb, WVb, bKp, bVp, Kb, Vb);
    init_kernel<<<BATCH * HID / 256, 256, 0, stream>>>(q, x, out);

    for (int t = 0; t < NUMX - 1; ++t) {
        scores_softmax_kernel<<<BATCH, 512, 0, stream>>>(Kb, x, attn);
        attend_v_kernel<<<dim3(2, BATCH), 256, 0, stream>>>(Vb, attn, x, out, t);
    }
}
